// ARNN_17188459118642
// MI455X (gfx1250) — matrix operand order repaired and checked by reading, not compiled, not run on hardware
//
#include <hip/hip_runtime.h>
#include <hip/hip_bf16.h>

// Problem constants (match reference)
#define B_  8
#define N_  2048
#define D_  128
#define H_  128
#define G4  512   // 4*H

typedef __attribute__((ext_vector_type(16))) _Float16 v16h;
typedef __attribute__((ext_vector_type(8)))  _Float16 h8;   // 16-byte half vector
typedef __attribute__((ext_vector_type(8)))  float    v8f;
typedef __attribute__((ext_vector_type(4)))  int      i4;   // native vector for NT loads

static __device__ __forceinline__ v8f zero8() {
  v8f r;
#pragma unroll
  for (int i = 0; i < 8; ++i) r[i] = 0.0f;
  return r;
}

static __device__ __forceinline__ float sigm(float x) {
  return 1.0f / (1.0f + __expf(-x));
}

// B-fragment (32x16 f16): lane holds column n = lane&15; half h -> K = h + (h<8?0:8) + (lane<16?0:8).
// Source must have the 32 K-values contiguous (row-major weight row, or transposed x column).
static __device__ __forceinline__ v16h load_bfrag_row(const _Float16* src_k0, int lane) {
  const int off = (lane & 16) ? 8 : 0;
  h8 a0 = *reinterpret_cast<const h8*>(src_k0 + off);
  h8 a1 = *reinterpret_cast<const h8*>(src_k0 + off + 16);
  v16h r;
#pragma unroll
  for (int i = 0; i < 8; ++i) { r[i] = a0[i]; r[i + 8] = a1[i]; }
  return r;
}

// A-fragment (16x32 f16): lane holds row m = lane&15; half h -> K = h + (h<8?0:8) + (lane<16?0:8).
// Source row-major: two contiguous 8-half groups.
static __device__ __forceinline__ v16h load_afrag_row(const _Float16* arow_k0, int lane) {
  const int off = (lane & 16) ? 8 : 0;
  h8 a0 = *reinterpret_cast<const h8*>(arow_k0 + off);
  h8 a1 = *reinterpret_cast<const h8*>(arow_k0 + 16 + off);
  v16h r;
#pragma unroll
  for (int i = 0; i < 8; ++i) { r[i] = a0[i]; r[i + 8] = a1[i]; }
  return r;
}

// ---------------------------------------------------------------------------
// Kernel 0: f32 -> f16 conversions.
//  - x is stored TRANSPOSED per batch: xhT[b][d][n], so aggregation B-fragments
//    become two contiguous b128 loads instead of 16 strided u16 loads.
//  - weights converted 1:1 (their fragments are already contiguous).
// ---------------------------------------------------------------------------
__global__ void k_convert(const float* __restrict__ x,
                          const float* __restrict__ wihf, const float* __restrict__ whhf,
                          const float* __restrict__ wihb, const float* __restrict__ whhb,
                          _Float16* __restrict__ xhT,
                          _Float16* __restrict__ hwihf, _Float16* __restrict__ hwhhf,
                          _Float16* __restrict__ hwihb, _Float16* __restrict__ hwhhb) {
  const int NX = B_ * N_ * D_;   // 2,097,152
  const int NW = G4 * D_;        // 65,536 (w_hh has same size since H == D)
  int idx = blockIdx.x * 256 + threadIdx.x;
  if (idx < NX) {
    // idx walks x row-major: b, n, d  (coalesced read)
    const int d = idx & (D_ - 1);
    const int n = (idx >> 7) & (N_ - 1);
    const int b = idx >> 18;
    xhT[((size_t)b * D_ + d) * N_ + n] = (_Float16)x[idx];
    return;
  }
  int j = idx - NX;
  if (j < NW) { hwihf[j] = (_Float16)wihf[j]; return; }
  j -= NW;
  if (j < NW) { hwhhf[j] = (_Float16)whhf[j]; return; }
  j -= NW;
  if (j < NW) { hwihb[j] = (_Float16)wihb[j]; return; }
  j -= NW;
  if (j < NW) { hwhhb[j] = (_Float16)whhb[j]; }
}

// ---------------------------------------------------------------------------
// Kernel 1: graph aggregation  xa = (x + a@x) / (1 + deg), output in f16.
// One wave per 16-node row tile; 8 WMMA accumulator tiles cover D=128.
// Adjacency streamed NON-TEMPORAL (134 MB touched once) so the reused x^T
// slabs stay resident in L2/WGP$; adjacency prefetched one K-step ahead;
// B-fragments kept as one live array (waves_per_eu(2) relaxes the register
// budget so the scheduler can batch all 16 b128 loads per K-step).
// ---------------------------------------------------------------------------
struct Adj4 { i4 a, b, c, d; };

static __device__ __forceinline__ Adj4 load_adj_nt(const int* __restrict__ adjp, int k0, int aoff) {
  Adj4 r;
  const i4* p0 = reinterpret_cast<const i4*>(adjp + k0 + aoff);
  const i4* p1 = reinterpret_cast<const i4*>(adjp + k0 + 16 + aoff);
  r.a = __builtin_nontemporal_load(p0);
  r.b = __builtin_nontemporal_load(p0 + 1);
  r.c = __builtin_nontemporal_load(p1);
  r.d = __builtin_nontemporal_load(p1 + 1);
  return r;
}

__global__ void __attribute__((amdgpu_waves_per_eu(2)))
k_aggregate(const int* __restrict__ adj, const _Float16* __restrict__ xhT,
            const float* __restrict__ x, _Float16* __restrict__ xah) {
  const int lane = threadIdx.x & 31;
  const int bt = blockIdx.x;            // 0..1023
  const int b  = bt >> 7;               // batch
  const int mt = bt & 127;              // 16-row tile
  const int row = mt * 16 + (lane & 15);
  const int* adjp = adj + ((size_t)b * N_ + row) * N_;
  const int aoff = (lane & 16) ? 8 : 0;
  const int d0 = lane & 15;
  const _Float16* xcol0 = xhT + (size_t)b * D_ * N_ + (size_t)d0 * N_;

  v8f acc[8];
#pragma unroll
  for (int t = 0; t < 8; ++t) acc[t] = zero8();
  float degp = 0.0f;

  Adj4 q = load_adj_nt(adjp, 0, aoff);  // prologue: adjacency for kt=0
  for (int kt = 0; kt < N_ / 32; ++kt) {
    const int k0 = kt * 32;
    // prefetch adjacency for the next step (last one is a harmless reload)
    const int kn = (kt + 1 < N_ / 32) ? (kt + 1) * 32 : k0;
    Adj4 qn = load_adj_nt(adjp, kn, aoff);

    // all 8 B fragments as one live array -> one clause of 16 b128 loads
    v16h bf[8];
#pragma unroll
    for (int t = 0; t < 8; ++t)
      bf[t] = load_bfrag_row(xcol0 + (size_t)t * 16 * N_ + k0, lane);

    // A fragment from the previously-loaded adjacency (no load wait here)
    int av[16] = { q.a[0], q.a[1], q.a[2], q.a[3], q.b[0], q.b[1], q.b[2], q.b[3],
                   q.c[0], q.c[1], q.c[2], q.c[3], q.d[0], q.d[1], q.d[2], q.d[3] };
    float av01[16];
    v16h af;
#pragma unroll
    for (int i = 0; i < 16; ++i) {
      av01[i] = (av[i] > 0) ? 1.0f : 0.0f;
      af[i] = (_Float16)av01[i];
    }
    // pairwise tree reduction for the degree (depth 4 instead of 16)
#pragma unroll
    for (int s = 1; s < 16; s <<= 1)
#pragma unroll
      for (int i = 0; i < 16; i += 2 * s) av01[i] += av01[i + s];
    degp += av01[0];

#pragma unroll
    for (int t = 0; t < 8; ++t)
      acc[t] = __builtin_amdgcn_wmma_f32_16x16x32_f16(false, af, false, bf[t],
                                                      (short)0, acc[t], false, false);
    q = qn;   // loop-carried SSA value; no register rotation copies
  }

  // Full row degree: lane and lane^16 hold disjoint K halves of the same row.
  float deg = degp + __shfl_xor(degp, 16, 32);
  const int mbase = (lane & 16) ? 8 : 0;
  float inv[8];
#pragma unroll
  for (int r = 0; r < 8; ++r)
    inv[r] = 1.0f / (1.0f + __shfl(deg, r + mbase, 32));  // lane m holds row m

  const float* xb = x + (size_t)b * N_ * D_;
  _Float16* xab = xah + (size_t)b * N_ * D_;
#pragma unroll
  for (int t = 0; t < 8; ++t) {
    const int d = t * 16 + d0;
#pragma unroll
    for (int r = 0; r < 8; ++r) {
      const int node = mt * 16 + r + mbase;
      const float xv = xb[(size_t)node * D_ + d];
      xab[(size_t)node * D_ + d] = (_Float16)((xv + acc[t][r]) * inv[r]);
    }
  }
}

// ---------------------------------------------------------------------------
// Kernel 2: input projection xp[t][bb][g] = xa[bb][node(t)] @ w_ih^T + (b_ih+b_hh)
// Rows flattened as t*8+bb; backward direction stores already-reversed sequence.
// 4 waves per block, each wave owns 8 gate-column tiles (128 columns).
// A-fragments loaded up front; one live bf[8] array per (unrolled) K step.
// ---------------------------------------------------------------------------
__global__ void k_proj(const _Float16* __restrict__ xah,
                       const _Float16* __restrict__ wfih, const _Float16* __restrict__ wbih,
                       const float* __restrict__ bihf, const float* __restrict__ bhhf,
                       const float* __restrict__ bihb, const float* __restrict__ bhhb,
                       float* __restrict__ xpf, float* __restrict__ xpb) {
  const int dir  = blockIdx.x >> 10;
  const int mt   = blockIdx.x & 1023;    // 16-row tile of the 16384 flattened rows
  const int wave = threadIdx.x >> 5;
  const int lane = threadIdx.x & 31;

  const _Float16* wih = dir ? wbih : wfih;
  const float* bih = dir ? bihb : bihf;
  const float* bhh = dir ? bhhb : bhhf;
  float* xp = dir ? xpb : xpf;

  const int grow = mt * 16 + (lane & 15);          // A-fragment row for this lane
  const int t  = grow >> 3;
  const int bb = grow & 7;
  const int node = dir ? (N_ - 1 - t) : t;
  const _Float16* arow = xah + ((size_t)bb * N_ + node) * D_;
  const _Float16* wrow = wih + (size_t)(wave * 128 + (lane & 15)) * D_;

  v8f acc[8];
#pragma unroll
  for (int j = 0; j < 8; ++j) acc[j] = zero8();

  // All 4 A-fragments up front (8 b128 loads)
  v16h af[4];
#pragma unroll
  for (int kk = 0; kk < 4; ++kk) af[kk] = load_afrag_row(arow + kk * 32, lane);

#pragma unroll
  for (int kk = 0; kk < 4; ++kk) {
    v16h bf[8];
#pragma unroll
    for (int j = 0; j < 8; ++j)
      bf[j] = load_bfrag_row(wrow + (size_t)(j * 16) * D_ + kk * 32, lane);
#pragma unroll
    for (int j = 0; j < 8; ++j)
      acc[j] = __builtin_amdgcn_wmma_f32_16x16x32_f16(false, af[kk], false, bf[j],
                                                      (short)0, acc[j], false, false);
  }

  const int mbase = (lane & 16) ? 8 : 0;
#pragma unroll
  for (int j = 0; j < 8; ++j) {
    const int g = (wave * 8 + j) * 16 + (lane & 15);
    const float bias = bih[g] + bhh[g];
#pragma unroll
    for (int r = 0; r < 8; ++r) {
      const int rr = mt * 16 + r + mbase;             // flattened row t*8+bb
      xp[(size_t)rr * G4 + g] = acc[j][r] + bias;
    }
  }
}

// ---------------------------------------------------------------------------
// Kernel 3: LSTM recurrence, one workgroup per direction (8 waves).
// w_hh fragments preloaded in VGPRs; h (f16, zero-padded to 16 rows) and c in LDS.
// Per step: 16 WMMAs/wave, xp add, gate nonlinearity, two barriers.
// ---------------------------------------------------------------------------
__global__ void __launch_bounds__(256) k_lstm(const float* __restrict__ xpf,
                                              const float* __restrict__ xpb,
                                              const _Float16* __restrict__ wfhh,
                                              const _Float16* __restrict__ wbhh,
                                              float* __restrict__ out) {
  __shared__ __align__(16) _Float16 hbuf[16 * H_];   // rows 8..15 stay zero
  __shared__ float gbuf[8 * G4];
  __shared__ float cbuf[8 * H_];

  const int dir  = blockIdx.x;
  const int tid  = threadIdx.x;
  const int wave = tid >> 5;
  const int lane = tid & 31;
  const float* xp = dir ? xpb : xpf;
  const _Float16* whh = dir ? wbhh : wfhh;

  for (int i = tid; i < 16 * H_; i += 256) hbuf[i] = (_Float16)0.0f;
  for (int i = tid; i < 8 * H_; i += 256) cbuf[i] = 0.0f;

  // Preload w_hh B-fragments: wave owns gate columns [wave*64, wave*64+64).
  v16h bw[4][4];
#pragma unroll
  for (int j = 0; j < 4; ++j) {
    const int g = (wave * 4 + j) * 16 + (lane & 15);
#pragma unroll
    for (int kk = 0; kk < 4; ++kk)
      bw[j][kk] = load_bfrag_row(whh + (size_t)g * H_ + kk * 32, lane);
  }
  __syncthreads();

  const int arow = (lane & 15) * H_;
  for (int t = 0; t < N_; ++t) {
    v16h af[4];
#pragma unroll
    for (int kk = 0; kk < 4; ++kk)
      af[kk] = load_afrag_row(&hbuf[arow + kk * 32], lane);

#pragma unroll
    for (int j = 0; j < 4; ++j) {
      v8f acc = zero8();
#pragma unroll
      for (int kk = 0; kk < 4; ++kk)
        acc = __builtin_amdgcn_wmma_f32_16x16x32_f16(false, af[kk], false, bw[j][kk],
                                                     (short)0, acc, false, false);
      if (lane < 16) {   // valid batch rows 0..7 live in lanes 0..15 (M = r)
        const int col = (wave * 4 + j) * 16 + lane;
        const float* xrow = xp + (size_t)t * (8 * G4) + col;
#pragma unroll
        for (int r = 0; r < 8; ++r)
          gbuf[r * G4 + col] = acc[r] + xrow[(size_t)r * G4];
      }
    }
    __syncthreads();

    for (int it = tid; it < 8 * H_; it += 256) {
      const int bb = it >> 7;
      const int hh = it & (H_ - 1);
      const float gi = gbuf[bb * G4 + hh];
      const float gf = gbuf[bb * G4 + H_ + hh];
      const float gg = gbuf[bb * G4 + 2 * H_ + hh];
      const float go = gbuf[bb * G4 + 3 * H_ + hh];
      const float c = sigm(gf) * cbuf[it] + sigm(gi) * tanhf(gg);
      const float h = sigm(go) * tanhf(c);
      cbuf[it] = c;
      hbuf[bb * H_ + hh] = (_Float16)h;
      if (t == N_ - 1) out[bb * (2 * H_) + dir * H_ + hh] = h;
    }
    __syncthreads();
  }
}

// ---------------------------------------------------------------------------
extern "C" void kernel_launch(void* const* d_in, const int* in_sizes, int n_in,
                              void* d_out, int out_size, void* d_ws, size_t ws_size,
                              hipStream_t stream) {
  const float* x      = (const float*)d_in[0];
  const int*   adj    = (const int*)d_in[1];
  const float* w_ih_f = (const float*)d_in[2];
  const float* w_hh_f = (const float*)d_in[3];
  const float* b_ih_f = (const float*)d_in[4];
  const float* b_hh_f = (const float*)d_in[5];
  const float* w_ih_b = (const float*)d_in[6];
  const float* w_hh_b = (const float*)d_in[7];
  const float* b_ih_b = (const float*)d_in[8];
  const float* b_hh_b = (const float*)d_in[9];
  float* out = (float*)d_out;

  // Workspace layout (~72.5 MB total)
  _Float16* xhT  = (_Float16*)d_ws;           // B*D*N f16 (transposed per batch)
  _Float16* xah  = xhT + (size_t)B_ * N_ * D_;
  _Float16* wfih = xah + (size_t)B_ * N_ * D_;
  _Float16* wfhh = wfih + (size_t)G4 * D_;
  _Float16* wbih = wfhh + (size_t)G4 * D_;
  _Float16* wbhh = wbih + (size_t)G4 * D_;
  float* xpf = (float*)(wbhh + (size_t)G4 * D_);  // [N][B][4H] f32 (time-major)
  float* xpb = xpf + (size_t)N_ * B_ * G4;        // already-reversed sequence

  const int NX = B_ * N_ * D_;
  const int NW = G4 * D_;
  const int conv_total = NX + 4 * NW;
  k_convert<<<(conv_total + 255) / 256, 256, 0, stream>>>(
      x, w_ih_f, w_hh_f, w_ih_b, w_hh_b, xhT, wfih, wfhh, wbih, wbhh);

  k_aggregate<<<B_ * (N_ / 16), 32, 0, stream>>>(adj, xhT, x, xah);

  k_proj<<<2 * ((N_ * B_) / 16), 128, 0, stream>>>(
      xah, wfih, wbih, b_ih_f, b_hh_f, b_ih_b, b_hh_b, xpf, xpb);

  k_lstm<<<2, 256, 0, stream>>>(xpf, xpb, wfhh, wbhh, out);
}
